// MultiScaleRetention_29721173688875
// MI455X (gfx1250) — matrix operand order repaired and checked by reading, not compiled, not run on hardware
//
#include <hip/hip_runtime.h>
#include <hip/hip_bf16.h>

#define BB 2
#define SS 2048
#define DD 2048
#define HH 16
#define HD 128

typedef __attribute__((ext_vector_type(16))) __bf16 v16bf;
typedef __attribute__((ext_vector_type(8)))  float  v8f;

// ---------- helpers ----------

__device__ __forceinline__ __bf16 f2bf(float f) {
  return (__bf16)f;   // native cast -> hardware v_cvt (RNE)
}

__device__ __forceinline__ v8f zero8() {
  v8f z;
#pragma unroll
  for (int i = 0; i < 8; ++i) z[i] = 0.0f;
  return z;
}

__device__ __forceinline__ v8f wmma_bf16(v16bf a, v16bf b, v8f c) {
  // D = A(16x32 bf16) x B(32x16 bf16) + C(16x16 f32)
  return __builtin_amdgcn_wmma_f32_16x16x32_bf16(false, a, false, b, (short)0, c,
                                                 false, false);
}

// A fragment (16x32, bf16) from row-major bf16 src (ld halves per row).
// Lane L holds row L%16; K = (L>=16 ? 8 : 0) + {0..7, 16..23}.
__device__ __forceinline__ v16bf load_a_bf16(const __bf16* src, int ld, int lane) {
  int r  = lane & 15;
  int kb = (lane & 16) ? 8 : 0;
  const __bf16* p = src + (long)r * ld + kb;
  union { v16bf v; uint4 q[2]; } u;
  u.q[0] = *(const uint4*)(p);
  u.q[1] = *(const uint4*)(p + 16);
  return u.v;
}

// B fragment (32x16) from B-transposed bf16 storage BT[n][k] (k contiguous).
// Lane L holds column L%16; K = (L>=16 ? 8 : 0) + {0..7, 16..23}.
__device__ __forceinline__ v16bf load_b_bf16T(const __bf16* bt, int ld, int lane) {
  int n  = lane & 15;
  int kb = (lane & 16) ? 8 : 0;
  const __bf16* p = bt + (long)n * ld + kb;
  union { v16bf v; uint4 q[2]; } u;
  u.q[0] = *(const uint4*)(p);
  u.q[1] = *(const uint4*)(p + 16);
  return u.v;
}

// ---------- kernel 0a: elementwise fp32 -> bf16 (8 elems/thread) ----------
__global__ void cvt_kernel(const float* __restrict__ in, __bf16* __restrict__ out,
                           long total8) {
  long i = (long)blockIdx.x * blockDim.x + threadIdx.x;
  if (i >= total8) return;
  const float4* p = (const float4*)in + i * 2;
  float4 f0 = p[0], f1 = p[1];
  union { uint4 u; __bf16 e[8]; } pk;
  pk.e[0] = f2bf(f0.x); pk.e[1] = f2bf(f0.y);
  pk.e[2] = f2bf(f0.z); pk.e[3] = f2bf(f0.w);
  pk.e[4] = f2bf(f1.x); pk.e[5] = f2bf(f1.y);
  pk.e[6] = f2bf(f1.z); pk.e[7] = f2bf(f1.w);
  ((uint4*)out)[i] = pk.u;
}

// ---------- kernel 0b: transpose + fp32->bf16 convert (weights) ----------
// in:  [batch][R][C] fp32   out: [batch][C][R] bf16
__global__ void tconv_kernel(const float* __restrict__ in, __bf16* __restrict__ out,
                             int R, int C, long total) {
  long i = (long)blockIdx.x * blockDim.x + threadIdx.x;
  if (i >= total) return;
  int  c  = (int)(i % C);
  long t  = i / C;
  int  r  = (int)(t % R);
  long bi = t / R;
  out[bi * (long)R * C + (long)c * R + r] = f2bf(in[i]);
}

// ---------- kernel 1: per-head q/k/v projection ----------
// one wave per (b,h,16-row s-tile); q,k row-major bf16 [b,h,s,d]; v transposed [b,h,d,s]
__global__ __launch_bounds__(256) void proj_kernel(
    const __bf16* __restrict__ xbf,
    const __bf16* __restrict__ WqT, const __bf16* __restrict__ WkT,
    const __bf16* __restrict__ WvT,
    __bf16* __restrict__ q, __bf16* __restrict__ k, __bf16* __restrict__ vT) {
  int lane = threadIdx.x & 31;
  int wv   = blockIdx.x * 8 + (threadIdx.x >> 5);
  int nb = wv & 127;          // S/16
  int t1 = wv >> 7;
  int h  = t1 & 15;           // H
  int b  = t1 >> 4;
  int s0 = nb * 16;

  const __bf16* xrow = xbf + ((long)b * SS + s0) * DD + h * HD;
  v16bf xa[4];
#pragma unroll
  for (int kk = 0; kk < 4; ++kk) xa[kk] = load_a_bf16(xrow + 32 * kk, DD, lane);

  const __bf16* Ws[3] = { WqT + (long)h * HD * HD,
                          WkT + (long)h * HD * HD,
                          WvT + (long)h * HD * HD };
  int N  = lane & 15;
  int Mb = (lane & 16) ? 8 : 0;
  long headoff = (long)(b * HH + h) * SS;

#pragma unroll
  for (int w = 0; w < 3; ++w) {
#pragma unroll
    for (int t = 0; t < 8; ++t) {
      v8f c = zero8();
#pragma unroll
      for (int kk = 0; kk < 4; ++kk)
        c = wmma_bf16(xa[kk],
                      load_b_bf16T(Ws[w] + (long)(16 * t) * HD + 32 * kk, HD, lane), c);
      if (w < 2) {
        __bf16* dst = (w == 0 ? q : k) + (headoff + s0) * HD + 16 * t + N;
#pragma unroll
        for (int r = 0; r < 8; ++r) dst[(long)(Mb + r) * HD] = f2bf(c[r]);
      } else {
        union { uint4 u; __bf16 e[8]; } pk;
#pragma unroll
        for (int r = 0; r < 8; ++r) pk.e[r] = f2bf(c[r]);
        __bf16* dst = vT + ((long)(b * HH + h) * HD + 16 * t + N) * SS + s0 + Mb;
        *(uint4*)dst = pk.u;
      }
    }
  }
}

// ---------- kernel 2: fused causal retention with decay ----------
// one wave per (b,h,16-row n-tile). O(16xHD) accumulated over causal m-tiles of 32.
__global__ __launch_bounds__(256) void retention_kernel(
    const __bf16* __restrict__ q, const __bf16* __restrict__ k,
    const __bf16* __restrict__ vT, float* __restrict__ ret) {
  __shared__ __bf16 scl[8][16 * 32];   // per-wave 16x32 bf16 score tile

  int lane = threadIdx.x & 31;
  int wid  = threadIdx.x >> 5;
  int wv   = blockIdx.x * 8 + wid;
  int nb = wv & 127;
  int t1 = wv >> 7;
  int h  = t1 & 15;
  int b  = t1 >> 4;
  int n0 = nb * 16;

  float lg512 = logf(1.0f / 512.0f), lg32 = logf(1.0f / 32.0f);
  float gamma = 1.0f - expf(lg512 + (lg32 - lg512) * ((float)h / (float)(HH - 1)));
  float l2g   = log2f(gamma);

  long  hbase = (long)(b * HH + h);
  const __bf16* qrow  = q  + (hbase * SS + n0) * HD;
  const __bf16* kbase = k  + hbase * SS * HD;
  const __bf16* vbase = vT + hbase * HD * SS;

  v16bf qa[4];
#pragma unroll
  for (int kk = 0; kk < 4; ++kk) qa[kk] = load_a_bf16(qrow + 32 * kk, HD, lane);

  v8f o[8];
#pragma unroll
  for (int t = 0; t < 8; ++t) o[t] = zero8();

  int N  = lane & 15;
  int Mb = (lane & 16) ? 8 : 0;

  for (int m0 = 0; m0 < n0 + 16; m0 += 32) {
#pragma unroll
    for (int hf = 0; hf < 2; ++hf) {
      int mm = m0 + 16 * hf;
      v8f c = zero8();
      if (mm < n0 + 16) {              // wave-uniform
        const __bf16* krow = kbase + (long)mm * HD;
#pragma unroll
        for (int kk = 0; kk < 4; ++kk)
          c = wmma_bf16(qa[kk], load_b_bf16T(krow + 32 * kk, HD, lane), c);
      }
      int m = mm + N;
#pragma unroll
      for (int r = 0; r < 8; ++r) {
        int n  = n0 + Mb + r;
        int dd = n - m;
        float wgt = (dd >= 0) ? exp2f((float)dd * l2g) : 0.0f;  // causal decay
        scl[wid][(Mb + r) * 32 + 16 * hf + N] = f2bf(c[r] * wgt);
      }
    }
    asm volatile("s_wait_dscnt 0" ::: "memory");   // wave-local LDS RAW fence
    v16bf sa = load_a_bf16(&scl[wid][0], 32, lane);
#pragma unroll
    for (int t = 0; t < 8; ++t) {
      v16bf vb = load_b_bf16T(vbase + (long)(16 * t) * SS + m0, SS, lane);
      o[t] = wmma_bf16(sa, vb, o[t]);
    }
  }

  // ret laid out [b, s, h, d] fp32
#pragma unroll
  for (int t = 0; t < 8; ++t)
#pragma unroll
    for (int r = 0; r < 8; ++r)
      ret[((long)b * SS + n0 + Mb + r) * DD + h * HD + 16 * t + N] = o[t][r];
}

// ---------- kernel 3: g = x @ W1, then swish, fp32 out ----------
// one wave computes 16 rows x 128 cols (8 accumulators)
__global__ __launch_bounds__(256) void gemm_swish_kernel(
    const __bf16* __restrict__ A, const __bf16* __restrict__ BT, float* __restrict__ Co) {
  int lane = threadIdx.x & 31;
  int wv   = blockIdx.x * 8 + (threadIdx.x >> 5);
  int ct = wv & 15;            // DD/128 col tiles
  int rt = wv >> 4;            // (B*S)/16 row tiles
  int r0 = rt * 16, c0 = ct * 128;
  v8f acc[8];
#pragma unroll
  for (int j = 0; j < 8; ++j) acc[j] = zero8();
  for (int k0 = 0; k0 < DD; k0 += 32) {
    v16bf a = load_a_bf16(A + (long)r0 * DD + k0, DD, lane);
#pragma unroll
    for (int j = 0; j < 8; ++j)
      acc[j] = wmma_bf16(a, load_b_bf16T(BT + (long)(c0 + 16 * j) * DD + k0, DD, lane),
                         acc[j]);
  }
  int N = lane & 15, Mb = (lane & 16) ? 8 : 0;
#pragma unroll
  for (int j = 0; j < 8; ++j)
#pragma unroll
    for (int r = 0; r < 8; ++r) {
      float g = acc[j][r];
      Co[(long)(r0 + Mb + r) * DD + c0 + 16 * j + N] = g / (1.0f + __expf(-g));
    }
}

// ---------- kernel 4: GroupNorm(ret) + swish(g), convert to bf16 ----------
// one wave per (b,s,h): HD=128 channels, 4 per lane.
__global__ __launch_bounds__(256) void gn_add_kernel(
    const float* __restrict__ ret, const float* __restrict__ tsw,
    const float* __restrict__ gn_w, const float* __restrict__ gn_b,
    __bf16* __restrict__ tbf) {
  int lane = threadIdx.x & 31;
  int wv   = blockIdx.x * 8 + (threadIdx.x >> 5);   // [0, B*S*H)
  int  h  = wv & 15;
  long bs = wv >> 4;                                // [0, B*S)
  const float* rp = ret + bs * DD + h * HD;
  float vals[4], s = 0.0f, sq = 0.0f;
#pragma unroll
  for (int j = 0; j < 4; ++j) {
    float v = rp[lane + 32 * j];
    vals[j] = v; s += v; sq += v * v;
  }
#pragma unroll
  for (int m = 16; m >= 1; m >>= 1) {
    s  += __shfl_xor(s, m, 32);
    sq += __shfl_xor(sq, m, 32);
  }
  float mean = s * (1.0f / HD);
  float var  = sq * (1.0f / HD) - mean * mean;
  float rstd = rsqrtf(var + 1e-5f);
  const float* tp = tsw + bs * DD + h * HD;
  __bf16* op = tbf + bs * DD + h * HD;
#pragma unroll
  for (int j = 0; j < 4; ++j) {
    int i = lane + 32 * j;
    float rn = (vals[j] - mean) * rstd * gn_w[h * HD + i] + gn_b[h * HD + i];
    op[i] = f2bf(tp[i] + rn);
  }
}

// ---------- kernel 5: out = t @ W2 (fp32 out) ----------
__global__ __launch_bounds__(256) void gemm_out_kernel(
    const __bf16* __restrict__ A, const __bf16* __restrict__ BT, float* __restrict__ Co) {
  int lane = threadIdx.x & 31;
  int wv   = blockIdx.x * 8 + (threadIdx.x >> 5);
  int ct = wv & 15;
  int rt = wv >> 4;
  int r0 = rt * 16, c0 = ct * 128;
  v8f acc[8];
#pragma unroll
  for (int j = 0; j < 8; ++j) acc[j] = zero8();
  for (int k0 = 0; k0 < DD; k0 += 32) {
    v16bf a = load_a_bf16(A + (long)r0 * DD + k0, DD, lane);
#pragma unroll
    for (int j = 0; j < 8; ++j)
      acc[j] = wmma_bf16(a, load_b_bf16T(BT + (long)(c0 + 16 * j) * DD + k0, DD, lane),
                         acc[j]);
  }
  int N = lane & 15, Mb = (lane & 16) ? 8 : 0;
#pragma unroll
  for (int j = 0; j < 8; ++j)
#pragma unroll
    for (int r = 0; r < 8; ++r)
      Co[(long)(r0 + Mb + r) * DD + c0 + 16 * j + N] = acc[j][r];
}

// ---------- launch ----------

extern "C" void kernel_launch(void* const* d_in, const int* in_sizes, int n_in,
                              void* d_out, int out_size, void* d_ws, size_t ws_size,
                              hipStream_t stream) {
  (void)in_sizes; (void)n_in; (void)out_size; (void)ws_size;
  const float* x   = (const float*)d_in[0];
  const float* Wq  = (const float*)d_in[1];
  const float* Wk  = (const float*)d_in[2];
  const float* Wv  = (const float*)d_in[3];
  const float* W1  = (const float*)d_in[4];
  const float* W2  = (const float*)d_in[5];
  const float* gnw = (const float*)d_in[6];
  const float* gnb = (const float*)d_in[7];
  float* out = (float*)d_out;

  char* ws = (char*)d_ws;
  auto alloc = [&](size_t bytes) -> void* {
    void* p = (void*)ws;
    ws += (bytes + 255) & ~(size_t)255;
    return p;
  };
  const long PH = (long)HH * HD * HD;          // per-weight head block
  const long QN = (long)BB * HH * SS * HD;     // q/k/v element count
  const long TN = (long)BB * SS * DD;          // B*S*D

  __bf16* WqT = (__bf16*)alloc(PH * 2);
  __bf16* WkT = (__bf16*)alloc(PH * 2);
  __bf16* WvT = (__bf16*)alloc(PH * 2);
  __bf16* W1T = (__bf16*)alloc((long)DD * DD * 2);
  __bf16* W2T = (__bf16*)alloc((long)DD * DD * 2);
  __bf16* xbf = (__bf16*)alloc(TN * 2);
  __bf16* qb  = (__bf16*)alloc(QN * 2);
  __bf16* kb  = (__bf16*)alloc(QN * 2);
  __bf16* vTb = (__bf16*)alloc(QN * 2);
  float*  ret = (float*) alloc(TN * 4);
  float*  tsw = (float*) alloc(TN * 4);
  __bf16* tbf = (__bf16*)alloc(TN * 2);

  // x -> bf16 (TN/8 threads)
  {
    long tot8 = TN / 8;
    cvt_kernel<<<(int)((tot8 + 255) / 256), 256, 0, stream>>>(x, xbf, tot8);
  }

  // weight transpose + convert
  {
    long tot = PH;
    int blk = (int)((tot + 255) / 256);
    tconv_kernel<<<blk, 256, 0, stream>>>(Wq, WqT, HD, HD, tot);
    tconv_kernel<<<blk, 256, 0, stream>>>(Wk, WkT, HD, HD, tot);
    tconv_kernel<<<blk, 256, 0, stream>>>(Wv, WvT, HD, HD, tot);
    long totW = (long)DD * DD;
    int blkW = (int)((totW + 255) / 256);
    tconv_kernel<<<blkW, 256, 0, stream>>>(W1, W1T, DD, DD, totW);
    tconv_kernel<<<blkW, 256, 0, stream>>>(W2, W2T, DD, DD, totW);
  }

  // qkv projection: B*H*(S/16) = 4096 waves -> 512 blocks x 8 waves
  proj_kernel<<<512, 256, 0, stream>>>(xbf, WqT, WkT, WvT, qb, kb, vTb);

  // fused causal retention: 4096 waves
  retention_kernel<<<512, 256, 0, stream>>>(qb, kb, vTb, ret);

  // swish(x @ W1): (4096/16)*(2048/128) = 4096 waves -> 512 blocks
  gemm_swish_kernel<<<512, 256, 0, stream>>>(xbf, W1T, tsw);

  // GroupNorm + add: B*S*H = 65536 waves -> 8192 blocks
  gn_add_kernel<<<8192, 256, 0, stream>>>(ret, tsw, gnw, gnb, tbf);

  // out = t @ W2
  gemm_out_kernel<<<512, 256, 0, stream>>>(tbf, W2T, out);
}
